// LXformer_40303973105907
// MI455X (gfx1250) — matrix operand order repaired and checked by reading, not compiled, not run on hardware
//
#include <hip/hip_runtime.h>
#include <hip/hip_bf16.h>

typedef __attribute__((ext_vector_type(16))) _Float16 v16h;
typedef __attribute__((ext_vector_type(8)))  float    v8f;

typedef unsigned int u32x4 __attribute__((ext_vector_type(4)));
typedef int          i32x8 __attribute__((ext_vector_type(8)));
typedef int          i32x4 __attribute__((ext_vector_type(4)));

#define NB   4
#define NP   8192
#define CCH  64
#define KNN  16
#define ROWS (NB*NP)          // 32768 global rows
#define LTC  192              // 3*C

// ---------------------------------------------------------------------------
// Kernel 1: brute-force 16-NN per query (3D coords). Candidate tiles (256 x
// float4 = 4KB) are staged into LDS by the Tensor Data Mover (one descriptor
// issued by wave 0, tracked with TENSORcnt), then all threads run the
// distance + register-resident sorted top-16 on that tile's data.
// ---------------------------------------------------------------------------
__global__ void knn_kernel(const float4* __restrict__ xytp4,
                           int* __restrict__ out_idx) {
    const int tid = threadIdx.x;                 // 0..255
    const int blk = blockIdx.x;                  // 0..127
    const int b   = blk >> 5;                    // 32 blocks per batch
    const int n   = ((blk & 31) << 8) + tid;     // query index within batch

    const float4 q = xytp4[b * NP + n];

    float bd[KNN];
    int   bi[KNN];
#pragma unroll
    for (int s = 0; s < KNN; ++s) { bd[s] = 3.4e38f; bi[s] = 0; }

    __shared__ float4 tile[256];

    for (int t0 = 0; t0 < NP; t0 += 256) {
        __syncthreads();                         // previous tile fully consumed
#if __has_builtin(__builtin_amdgcn_tensor_load_to_lds)
        if (tid < 32) {                          // wave 0 drives the TDM
            unsigned long long ga =
                (unsigned long long)(const void*)(xytp4 + (b * NP + t0));
            unsigned int lds = (unsigned int)(unsigned long long)(void*)&tile[0];
            // D# group0: count=1 | lds_addr | global_addr[56:0] | type=2
            u32x4 g0;
            g0[0] = 1u;                               // count=1, flags=0
            g0[1] = lds;                              // LDS byte address
            g0[2] = (unsigned int)(ga & 0xffffffffu); // global_addr[31:0]
            g0[3] = (unsigned int)((ga >> 32) & 0x01ffffffu) | (2u << 30); // type=2
            // D# group1: 2D tile 256 rows x 4 floats, dense (stride 4 elems)
            i32x8 g1;
            g1[0] = (2 << 16);        // data_size=2 (4 bytes), wg_mask=0
            g1[1] = (4 << 16);        // tensor_dim0 = 4 elements
            g1[2] = (256 << 16);      // tensor_dim1 = 256 rows
            g1[3] = (4 << 16);        // tile_dim0 = 4 elements
            g1[4] = 256;              // tile_dim1 = 256 rows, tile_dim2 = 0
            g1[5] = 4;                // tensor_dim0_stride = 4 elements
            g1[6] = 0;
            g1[7] = 0;
            i32x4 g2 = {0, 0, 0, 0};
            i32x4 g3 = {0, 0, 0, 0};
#if defined(__clang_major__) && (__clang_major__ >= 23)
            i32x8 g4 = {0, 0, 0, 0, 0, 0, 0, 0};
            __builtin_amdgcn_tensor_load_to_lds(g0, g1, g2, g3, g4, 0);
#else
            __builtin_amdgcn_tensor_load_to_lds(g0, g1, g2, g3, 0);
#endif
            __builtin_amdgcn_s_wait_tensorcnt(0);
        }
#else
        tile[tid] = xytp4[b * NP + t0 + tid];
#endif
        __syncthreads();                         // tile visible to all waves
        for (int j = 0; j < 256; ++j) {
            float4 r = tile[j];
            const float dx = q.x - r.x;
            const float dy = q.y - r.y;
            const float dz = q.z - r.z;
            float d2 = dx*dx + dy*dy + dz*dz;
            if (d2 < bd[KNN-1]) {                // rare path: sorted insert
                float cd = d2; int ci = t0 + j;
#pragma unroll
                for (int s = 0; s < KNN; ++s) {
                    bool sw = cd < bd[s];
                    float td = bd[s]; int ti = bi[s];
                    bd[s] = sw ? cd : td;  bi[s] = sw ? ci : ti;
                    cd    = sw ? td : cd;  ci    = sw ? ti : ci;
                }
            }
        }
    }
#pragma unroll
    for (int s = 0; s < KNN; ++s)
        out_idx[(b * NP + n) * KNN + s] = bi[s];
}

// ---------------------------------------------------------------------------
// Kernel 2: lt = features @ lt_w + lt_b   (32768x64 @ 64x192)
// One wave per 16x16 tile, K=64 as two v_wmma_f32_16x16x32_f16 steps.
// ---------------------------------------------------------------------------
__global__ void lt_gemm_kernel(const float* __restrict__ feat,
                               const float* __restrict__ lt_w,
                               const float* __restrict__ lt_b,
                               float* __restrict__ lt_out) {
    const int lane = threadIdx.x & 31;
    const int wv   = threadIdx.x >> 5;
    const int tile = blockIdx.x * 8 + wv;        // 24576 tiles total
    const int rt   = tile / 12;
    const int ct   = tile - rt * 12;
    const int r0   = rt << 4;
    const int n0   = ct << 4;
    const int Lm   = lane & 15;
    const int hi   = lane >> 4;

    const float* arow = feat + (r0 + Lm) * CCH;
    const int nn = n0 + Lm;

    v8f acc = {};
#pragma unroll
    for (int ks = 0; ks < 2; ++ks) {
        const int k0  = ks * 32;
        const int off = hi * 8;
        v16h a, bf;
#pragma unroll
        for (int j = 0; j < 8; ++j) {
            a[j]     = (_Float16)arow[k0 + off + j];
            a[8 + j] = (_Float16)arow[k0 + 16 + off + j];
        }
        const int joff = hi * 16;
#pragma unroll
        for (int j = 0; j < 16; ++j)
            bf[j] = (_Float16)lt_w[(k0 + (j >> 3) * 16 + hi * 8 + (j & 7)) * LTC + nn];
        acc = __builtin_amdgcn_wmma_f32_16x16x32_f16(
                  false, a, false, bf, (short)0, acc, false, false);
    }
    const float bias = lt_b[nn];
#pragma unroll
    for (int r = 0; r < 8; ++r) {
        const int M = r + hi * 8;
        lt_out[(r0 + M) * LTC + nn] = acc[r] + bias;
    }
}

// ---------------------------------------------------------------------------
// Kernel 3: fused delta-MLP (WMMA) + gather + layernorm(C) + softmax(K) + sum
// One wave per query point; delta lands in the WMMA C-layout:
// lane holds channel c = t*16+(lane&15), neighbor k = r + (lane>>4)*8.
// ---------------------------------------------------------------------------
__global__ void attn_kernel(const float4* __restrict__ xytp4,
                            const float*  __restrict__ pe_w1,
                            const float*  __restrict__ pe_b1,
                            const float*  __restrict__ pe_w2,
                            const float*  __restrict__ pe_b2,
                            const float*  __restrict__ ln_g,
                            const float*  __restrict__ ln_b,
                            const int*    __restrict__ knn,
                            const float*  __restrict__ lt,
                            float*        __restrict__ out) {
    __shared__ _Float16 sW2[4][2][32][16];   // pe_w2 pre-swizzled B-frags, 8 KB
    __shared__ float    sW1[4][CCH];         // pe_w1
    __shared__ float    sB1[CCH], sB2[CCH], sG[CCH], sBt[CCH];
    __shared__ _Float16 sH[8][16][CCH];      // per-wave h (16x64 f16), 16 KB
    __shared__ float    sRel[8][16][4];      // per-wave rel (16x4)

    const int tid  = threadIdx.x;
    const int lane = tid & 31;
    const int wv   = tid >> 5;
    const int Lm   = lane & 15;
    const int hi   = lane >> 4;

    // ---- stage shared weights (pe_w2 swizzled into per-lane B fragments) ----
    {
        const int t   = tid >> 6;
        const int ks  = (tid >> 5) & 1;
        const int ln2 = tid & 31;
        const int lm2 = ln2 & 15, h2 = ln2 >> 4;
        const int nn  = t * 16 + lm2;
        const int joff = h2 * 16, k0 = ks * 32;
#pragma unroll
        for (int j = 0; j < 16; ++j)
            sW2[t][ks][ln2][j] = (_Float16)pe_w2[(k0 + (j >> 3) * 16 + h2 * 8 + (j & 7)) * CCH + nn];
        sW1[tid >> 6][tid & 63] = pe_w1[tid];          // 256 elements exactly
        if (tid < CCH) {
            sB1[tid] = pe_b1[tid]; sB2[tid] = pe_b2[tid];
            sG[tid]  = ln_g[tid];  sBt[tid] = ln_b[tid];
        }
    }
    __syncthreads();

    const int q = blockIdx.x * 8 + wv;       // global row index == b*NP + n
    const int b = q >> 13;
    const int n = q & (NP - 1);

    const int myidx = knn[q * KNN + Lm];     // both half-waves load idx[Lm]

    // ---- rel[k][0..3] = xytp[n] - xytp[idx[k]] (full 4 dims) ----
    if (lane < 16) {
        float4 qc = xytp4[b * NP + n];
        float4 pc = xytp4[b * NP + myidx];
        sRel[wv][lane][0] = qc.x - pc.x;
        sRel[wv][lane][1] = qc.y - pc.y;
        sRel[wv][lane][2] = qc.z - pc.z;
        sRel[wv][lane][3] = qc.w - pc.w;
    }
    __syncthreads();

    // ---- h = relu(rel @ pe_w1 + pe_b1), 16x64, 32 elements per lane ----
#pragma unroll 4
    for (int i = 0; i < 32; ++i) {
        const int e = i * 32 + lane;
        const int k = e >> 6, c = e & 63;
        float v = sB1[c]
                + sRel[wv][k][0] * sW1[0][c]
                + sRel[wv][k][1] * sW1[1][c]
                + sRel[wv][k][2] * sW1[2][c]
                + sRel[wv][k][3] * sW1[3][c];
        sH[wv][k][c] = (_Float16)fmaxf(v, 0.0f);
    }
    __syncthreads();

    // ---- delta = h @ pe_w2 : 8 WMMAs (2 K-steps x 4 N-tiles) ----
    v8f acc[4] = {v8f{}, v8f{}, v8f{}, v8f{}};
#pragma unroll
    for (int ks = 0; ks < 2; ++ks) {
        const int k0 = ks * 32, off = hi * 8;
        v16h a;
#pragma unroll
        for (int j = 0; j < 8; ++j) {
            a[j]     = sH[wv][Lm][k0 + off + j];
            a[8 + j] = sH[wv][Lm][k0 + 16 + off + j];
        }
#pragma unroll
        for (int t = 0; t < 4; ++t) {
            v16h bf;
#pragma unroll
            for (int j = 0; j < 16; ++j) bf[j] = sW2[t][ks][lane][j];
            acc[t] = __builtin_amdgcn_wmma_f32_16x16x32_f16(
                         false, a, false, bf, (short)0, acc[t], false, false);
        }
    }

    // ---- gather psi/alpha, build pre (LN input) and s2 = alpha_g + delta ----
    float pre[4][8], s2[4][8];
#pragma unroll
    for (int t = 0; t < 4; ++t) {
        const int ct  = t * 16 + Lm;
        const float vphi = lt[q * LTC + ct];
        const float b2c  = sB2[ct];
#pragma unroll
        for (int r = 0; r < 8; ++r) {
            const int k = r + hi * 8;
            const int gidx = __shfl(myidx, k, 32);
            const float* grow = lt + (b * NP + gidx) * LTC;
            const float psi = grow[CCH + ct];
            const float alp = grow[2 * CCH + ct];
            const float d   = acc[t][r] + b2c;
            pre[t][r] = vphi - psi + d;
            s2[t][r]  = alp + d;
        }
    }

    // ---- layernorm over channels (reduce across 4 tiles + 16 lanes/half) ----
#pragma unroll
    for (int r = 0; r < 8; ++r) {
        float s = pre[0][r] + pre[1][r] + pre[2][r] + pre[3][r];
        s += __shfl_xor(s, 1, 32);  s += __shfl_xor(s, 2, 32);
        s += __shfl_xor(s, 4, 32);  s += __shfl_xor(s, 8, 32);
        const float mu = s * (1.0f / 64.0f);
        float d0 = pre[0][r] - mu, d1 = pre[1][r] - mu;
        float d2 = pre[2][r] - mu, d3 = pre[3][r] - mu;
        float ss = d0*d0 + d1*d1 + d2*d2 + d3*d3;
        ss += __shfl_xor(ss, 1, 32);  ss += __shfl_xor(ss, 2, 32);
        ss += __shfl_xor(ss, 4, 32);  ss += __shfl_xor(ss, 8, 32);
        const float rstd = rsqrtf(ss * (1.0f / 64.0f) + 1e-5f);
#pragma unroll
        for (int t = 0; t < 4; ++t) {
            const int ct = t * 16 + Lm;
            pre[t][r] = ((pre[t][r] - mu) * rstd * sG[ct] + sBt[ct]) * 0.125f;
        }
    }

    // ---- softmax over k (8 regs + cross-half) and weighted sum ----
    float o[4];
#pragma unroll
    for (int t = 0; t < 4; ++t) {
        float m = pre[t][0];
#pragma unroll
        for (int r = 1; r < 8; ++r) m = fmaxf(m, pre[t][r]);
        m = fmaxf(m, __shfl_xor(m, 16, 32));
        float den = 0.0f;
#pragma unroll
        for (int r = 0; r < 8; ++r) { float e = __expf(pre[t][r] - m); pre[t][r] = e; den += e; }
        den += __shfl_xor(den, 16, 32);
        const float inv = 1.0f / den;
        float part = 0.0f;
#pragma unroll
        for (int r = 0; r < 8; ++r) part += pre[t][r] * inv * s2[t][r];
        part += __shfl_xor(part, 16, 32);
        o[t] = part;
    }
    if (hi == 0) {
#pragma unroll
        for (int t = 0; t < 4; ++t)
            out[q * CCH + t * 16 + Lm] = o[t];
    }
}

// ---------------------------------------------------------------------------
extern "C" void kernel_launch(void* const* d_in, const int* in_sizes, int n_in,
                              void* d_out, int out_size, void* d_ws, size_t ws_size,
                              hipStream_t stream) {
    const float4* xytp4 = (const float4*)d_in[0];
    const float*  feat  = (const float*)d_in[1];
    const float*  pe_w1 = (const float*)d_in[2];
    const float*  pe_b1 = (const float*)d_in[3];
    const float*  pe_w2 = (const float*)d_in[4];
    const float*  pe_b2 = (const float*)d_in[5];
    const float*  lt_w  = (const float*)d_in[6];
    const float*  lt_b  = (const float*)d_in[7];
    const float*  ln_g  = (const float*)d_in[8];
    const float*  ln_b  = (const float*)d_in[9];
    float* out = (float*)d_out;

    // workspace layout: [idx: ROWS*16 ints = 2MB][lt: ROWS*192 floats = 24MB]
    int*   ws_idx = (int*)d_ws;
    float* ws_lt  = (float*)((char*)d_ws + (size_t)ROWS * KNN * sizeof(int));

    knn_kernel<<<(NB * NP) / 256, 256, 0, stream>>>(xytp4, ws_idx);

    // 2048 row-tiles * 12 col-tiles = 24576 tiles, 8 waves per 256-thread block
    lt_gemm_kernel<<<24576 / 8, 256, 0, stream>>>(feat, lt_w, lt_b, ws_lt);

    attn_kernel<<<ROWS / 8, 256, 0, stream>>>(xytp4, pe_w1, pe_b1, pe_w2, pe_b2,
                                              ln_g, ln_b, ws_idx, ws_lt, out);
}
